// Block_47467978555798
// MI455X (gfx1250) — matrix operand order repaired and checked by reading, not compiled, not run on hardware
//
#include <hip/hip_runtime.h>
#include <hip/hip_bf16.h>

// ---------------------------------------------------------------------------
// MI455X / gfx1250 transformer block: f16 WMMA GEMMs + flash attention.
// GEMMs: double-buffered LDS tiles filled with GLOBAL_LOAD_ASYNC_TO_LDS_B128
// (ASYNCcnt / s_wait_asynccnt), consumed by v_wmma_f32_16x16x32_f16 (wave32).
// ---------------------------------------------------------------------------

typedef __attribute__((ext_vector_type(16))) _Float16 v16h;
typedef __attribute__((ext_vector_type(8)))  float    v8f;
typedef __attribute__((ext_vector_type(4)))  int      v4i;
typedef __attribute__((ext_vector_type(2)))  int      v2i;

#define WMMA_F16(a, b, c) \
  __builtin_amdgcn_wmma_f32_16x16x32_f16(false, (a), false, (b), (short)0, (c), false, false)

// --- CDNA5 async global->LDS copy (16 bytes per lane) ---------------------
// Builtin signature (from hipcc diagnostic): param0 = v4i addrspace(1)* (non-
// const), param1 = v4i addrspace(3)*, then (offset, cpol) immediates.
#if __has_builtin(__builtin_amdgcn_global_load_async_to_lds_b128)
#define ASYNC_CP16(gsrc, ldst)                                                 \
  __builtin_amdgcn_global_load_async_to_lds_b128(                              \
      (__attribute__((address_space(1))) v4i*)(gsrc),                          \
      (__attribute__((address_space(3))) v4i*)(ldst), 0, 0)
#else
#define ASYNC_CP16(gsrc, ldst)                                                 \
  asm volatile("global_load_async_to_lds_b128 %0, %1, off"                     \
               :: "v"((unsigned)(unsigned long long)                           \
                        (__attribute__((address_space(3))) void*)(ldst)),      \
                  "v"((const void*)(gsrc))                                     \
               : "memory")
#endif

#if __has_builtin(__builtin_amdgcn_s_wait_asynccnt)
#define WAIT_ASYNC(n) __builtin_amdgcn_s_wait_asynccnt(n)
#else
#define WAIT_ASYNC(n) asm volatile("s_wait_asynccnt %0" :: "i"(n) : "memory")
#endif

static __device__ inline v8f zero8() {
  v8f z;
#pragma unroll
  for (int i = 0; i < 8; ++i) z[i] = 0.0f;
  return z;
}

// Model constants
constexpr int Bsz = 2, Sq = 2048, Dm = 1024, Hh = 16, HSz = 64;

// ---------------------------------------------------------------------------
// Weight conversion kernels (fp32 -> f16, transposed to [N][K] row-major)
// ---------------------------------------------------------------------------

// Wq/Wk/Wv: [H, D, HS] f32  ->  [N=H*HS][K=D] f16
__global__ __launch_bounds__(256) void cvt_qkvT(const float* __restrict__ in,
                                                _Float16* __restrict__ out) {
  int idx = blockIdx.x * 256 + threadIdx.x;          // over H*D*HS = 1M
  int h = idx >> 16;                                  // D*HS = 65536
  int r = idx & 65535;
  int d = r >> 6;
  int e = r & 63;
  out[(size_t)(h * HSz + e) * Dm + d] = (_Float16)in[idx];
}

// W: [K, N] f32 -> [N][K] f16
__global__ __launch_bounds__(256) void cvtT(const float* __restrict__ in,
                                            _Float16* __restrict__ out,
                                            int K, int N) {
  int idx = blockIdx.x * 256 + threadIdx.x;
  if (idx >= K * N) return;
  int k = idx / N;
  int n = idx - k * N;
  out[(size_t)n * K + k] = (_Float16)in[idx];
}

// ---------------------------------------------------------------------------
// LayerNorm over D=1024, one row per 256-thread block, f16 output
// ---------------------------------------------------------------------------
__global__ __launch_bounds__(256) void ln_kernel(const float* __restrict__ x,
                                                 const float* __restrict__ g,
                                                 const float* __restrict__ bb,
                                                 _Float16* __restrict__ out) {
  __shared__ float sP[8], sQ[8];
  const int row = blockIdx.x;
  const int t = threadIdx.x;
  const float* xr = x + (size_t)row * Dm;
  float4 v = *(const float4*)(xr + t * 4);
  float s1 = v.x + v.y + v.z + v.w;
  float s2 = v.x * v.x + v.y * v.y + v.z * v.z + v.w * v.w;
#pragma unroll
  for (int off = 16; off > 0; off >>= 1) {
    s1 += __shfl_xor(s1, off, 32);
    s2 += __shfl_xor(s2, off, 32);
  }
  if ((t & 31) == 0) { sP[t >> 5] = s1; sQ[t >> 5] = s2; }
  __syncthreads();
  float t1 = 0.f, t2 = 0.f;
#pragma unroll
  for (int i = 0; i < 8; ++i) { t1 += sP[i]; t2 += sQ[i]; }
  const float mu = t1 * (1.0f / Dm);
  const float var = t2 * (1.0f / Dm) - mu * mu;
  const float rstd = rsqrtf(var + 1e-5f);
  float4 gv = *(const float4*)(g + t * 4);
  float4 bv = *(const float4*)(bb + t * 4);
  union { _Float16 h[4]; v2i q; } u;
  u.h[0] = (_Float16)((v.x - mu) * rstd * gv.x + bv.x);
  u.h[1] = (_Float16)((v.y - mu) * rstd * gv.y + bv.y);
  u.h[2] = (_Float16)((v.z - mu) * rstd * gv.z + bv.z);
  u.h[3] = (_Float16)((v.w - mu) * rstd * gv.w + bv.w);
  *(v2i*)(out + (size_t)row * Dm + t * 4) = u.q;
}

// ---------------------------------------------------------------------------
// Generic f16 WMMA GEMM: C[M,N] = A[M,K] * Bt[N,K]^T, templated epilogue.
// Block = 256 threads (8 waves), tile 128x128x32, wave tile 64x32.
// Double-buffered LDS, filled by async global->LDS DMA (4 ops/wave/tile).
// ---------------------------------------------------------------------------
constexpr int EPI_Q = 0, EPI_K = 1, EPI_V = 2, EPI_PROJ = 3, EPI_FF1 = 4, EPI_FF2 = 5;

template <int EPI>
__global__ __launch_bounds__(256) void gemm_wmma(
    const _Float16* __restrict__ A, const _Float16* __restrict__ Bt,
    _Float16* outH, float* outF, const float* bias, const float* res,
    int M, int N, int K, float scale) {
  __shared__ __align__(16) _Float16 As[2][128 * 40];
  __shared__ __align__(16) _Float16 Bs[2][128 * 40];
  const int tid = threadIdx.x;
  const int lane = tid & 31, w = tid >> 5;
  const int wm = w >> 2, wn = w & 3;
  const int lm = lane & 15, hib = lane >> 4;
  const int kbA = hib ? 16 : 0;  // A-fragment k base per lane half
  const int kbB = hib ? 16 : 0;  // B-fragment k base per lane half
  const int m0 = blockIdx.y * 128, n0 = blockIdx.x * 128;
  const int fr = tid >> 1, fk = (tid & 1) * 16;  // LDS fill: row / k-chunk

  // Issue one 128x32 A tile + 128x32 B tile into LDS buffer `buf` (async).
  auto issue_tile = [&](int k0, int buf) {
    const _Float16* ga = A + (size_t)(m0 + fr) * K + k0 + fk;
    const _Float16* gb = Bt + (size_t)(n0 + fr) * K + k0 + fk;
    _Float16* la = &As[buf][fr * 40 + fk];
    _Float16* lb = &Bs[buf][fr * 40 + fk];
    ASYNC_CP16(ga, la);
    ASYNC_CP16(ga + 8, la + 8);
    ASYNC_CP16(gb, lb);
    ASYNC_CP16(gb + 8, lb + 8);
  };

  v8f acc[4][2];
#pragma unroll
  for (int a = 0; a < 4; ++a)
#pragma unroll
    for (int b = 0; b < 2; ++b) acc[a][b] = zero8();

  issue_tile(0, 0);
  const int nk = K >> 5;
  for (int it = 0; it < nk; ++it) {
    const int cur = it & 1;
    if (it + 1 < nk) {
      issue_tile((it + 1) * 32, cur ^ 1);
      WAIT_ASYNC(4);   // oldest 4 (current tile) complete; prefetch in flight
    } else {
      WAIT_ASYNC(0);
    }
    __syncthreads();   // all waves' tile data visible in LDS

    v16h bf[2];
#pragma unroll
    for (int tn = 0; tn < 2; ++tn) {
      const _Float16* r = &Bs[cur][(wn * 32 + tn * 16 + lm) * 40 + kbB];
      union { v16h v; v4i q[2]; } u;
      u.q[0] = *(const v4i*)r;
      u.q[1] = *(const v4i*)(r + 8);
      bf[tn] = u.v;
    }
#pragma unroll
    for (int tm = 0; tm < 4; ++tm) {
      const _Float16* r = &As[cur][(wm * 64 + tm * 16 + lm) * 40 + kbA];
      union { v16h v; v4i q[2]; } u;
      u.q[0] = *(const v4i*)r;
      u.q[1] = *(const v4i*)(r + 8);
      v16h af = u.v;
#pragma unroll
      for (int tn = 0; tn < 2; ++tn) acc[tm][tn] = WMMA_F16(af, bf[tn], acc[tm][tn]);
    }
    __syncthreads();   // protect buffer `cur` before it is refilled (WAR)
  }

  // Epilogue: C element (lane,i) -> m = i + 8*hi, n = lane%16 within each tile.
#pragma unroll
  for (int tm = 0; tm < 4; ++tm)
#pragma unroll
    for (int tn = 0; tn < 2; ++tn)
#pragma unroll
      for (int i = 0; i < 8; ++i) {
        const int m = m0 + wm * 64 + tm * 16 + i + (hib ? 8 : 0);
        const int n = n0 + wn * 32 + tn * 16 + lm;
        const float c = acc[tm][tn][i] * scale;
        if constexpr (EPI == EPI_Q || EPI == EPI_K) {
          const int b = m >> 11, s = m & 2047, h = n >> 6, e = n & 63;
          outH[((size_t)(b * Hh + h) * Sq + s) * HSz + e] = (_Float16)c;
        } else if constexpr (EPI == EPI_V) {
          const int b = m >> 11, s = m & 2047, h = n >> 6, e = n & 63;
          outH[((size_t)(b * Hh + h) * HSz + e) * Sq + s] = (_Float16)c;  // V^T
        } else if constexpr (EPI == EPI_PROJ) {
          const size_t idx = (size_t)m * N + n;
          outF[idx] = c + bias[n] + res[idx];
        } else if constexpr (EPI == EPI_FF1) {
          const float v = c + bias[n];
          outH[(size_t)m * N + n] = (_Float16)(v > 0.f ? v : 0.f);
        } else {  // EPI_FF2: residual in-place into d_out
          const size_t idx = (size_t)m * N + n;
          outF[idx] = res[idx] + c + bias[n];
        }
      }
}

// ---------------------------------------------------------------------------
// Flash attention: grid = (S/256, B*H); 8 waves/block; wave owns 32 q rows.
// Scores computed transposed (S^T = K x Q^T) so softmax stats live in-lane.
// P re-layout via per-wave LDS; O += P x V with V^T streamed from global.
// ---------------------------------------------------------------------------
__global__ __launch_bounds__(256) void attn_wmma(const _Float16* __restrict__ qg,
                                                 const _Float16* __restrict__ kg,
                                                 const _Float16* __restrict__ vTg,
                                                 _Float16* __restrict__ og) {
  __shared__ __align__(16) _Float16 Plds[8 * 2 * 16 * 40];
  const int lane = threadIdx.x & 31;
  const int w = threadIdx.x >> 5;
  const int bh = blockIdx.y;
  const int b = bh >> 4, h = bh & 15;
  const int sRow = blockIdx.x * 256 + w * 32;
  const _Float16* qp = qg + (size_t)bh * Sq * HSz;
  const _Float16* kp = kg + (size_t)bh * Sq * HSz;
  const _Float16* vp = vTg + (size_t)bh * HSz * Sq;
  const int lm = lane & 15, hib = lane >> 4;
  const int kbA = hib ? 8 : 0;
  const int kbB = hib ? 8 : 0;
  _Float16* Pw = Plds + w * (2 * 16 * 40);

  // Q^T B-fragments (held for whole pass): [msub][echunk]
  v16h Qb[2][2];
#pragma unroll
  for (int ms = 0; ms < 2; ++ms)
#pragma unroll
    for (int ec = 0; ec < 2; ++ec) {
      const _Float16* r = qp + (size_t)(sRow + ms * 16 + lm) * HSz + ec * 32 + kbB;
      union { v16h v; v4i q[2]; } u;
      u.q[0] = *(const v4i*)r;
      u.q[1] = *(const v4i*)(r + 16);
      Qb[ms][ec] = u.v;
    }

  v8f O[2][4];
#pragma unroll
  for (int a = 0; a < 2; ++a)
#pragma unroll
    for (int e = 0; e < 4; ++e) O[a][e] = zero8();
  float Mrow[2] = {-1e30f, -1e30f};
  float Lrow[2] = {0.f, 0.f};

  for (int t0 = 0; t0 < Sq; t0 += 32) {
    // --- scores transposed: S^T[t, m] = sum_e k[t,e] * q[m,e] (q pre-scaled) ---
    v8f Sv[2][2];  // [msub][tsub]
#pragma unroll
    for (int ts = 0; ts < 2; ++ts) {
      v16h Ka[2];
#pragma unroll
      for (int ec = 0; ec < 2; ++ec) {
        const _Float16* r = kp + (size_t)(t0 + ts * 16 + lm) * HSz + ec * 32 + kbA;
        union { v16h v; v4i q[2]; } u;
        u.q[0] = *(const v4i*)r;
        u.q[1] = *(const v4i*)(r + 16);
        Ka[ec] = u.v;
      }
#pragma unroll
      for (int ms = 0; ms < 2; ++ms) {
        v8f c = zero8();
        c = WMMA_F16(Ka[0], Qb[ms][0], c);
        c = WMMA_F16(Ka[1], Qb[ms][1], c);
        Sv[ms][ts] = c;
      }
    }
    // --- online softmax (stats keyed by column m = lane%16) ---
#pragma unroll
    for (int ms = 0; ms < 2; ++ms) {
      float cm = -1e30f;
#pragma unroll
      for (int ts = 0; ts < 2; ++ts)
#pragma unroll
        for (int i = 0; i < 8; ++i) cm = fmaxf(cm, Sv[ms][ts][i]);
      cm = fmaxf(cm, __shfl_xor(cm, 16, 32));
      const float newM = fmaxf(Mrow[ms], cm);
      const float alpha = __expf(Mrow[ms] - newM);
      Mrow[ms] = newM;
      float rs = 0.f;
      _Float16 ph[2][8];
#pragma unroll
      for (int ts = 0; ts < 2; ++ts)
#pragma unroll
        for (int i = 0; i < 8; ++i) {
          const float p = __expf(Sv[ms][ts][i] - newM);
          rs += p;
          ph[ts][i] = (_Float16)p;
        }
      rs += __shfl_xor(rs, 16, 32);
      Lrow[ms] = Lrow[ms] * alpha + rs;
      // rescale O rows (row index m = i + 8*hi; stats live at lane==m)
#pragma unroll
      for (int i = 0; i < 8; ++i) {
        const float af = __shfl(alpha, i + (hib ? 8 : 0), 32);
#pragma unroll
        for (int es = 0; es < 4; ++es) O[ms][es][i] *= af;
      }
      // stage P into per-wave LDS as row-major [m][t]
#pragma unroll
      for (int ts = 0; ts < 2; ++ts) {
        union { v4i q; _Float16 hh[8]; } u;
#pragma unroll
        for (int i = 0; i < 8; ++i) u.hh[i] = ph[ts][i];
        *(v4i*)(Pw + ms * 640 + lm * 40 + ts * 16 + kbA) = u.q;
      }
    }
    // --- O += P x V : A = P (from LDS), B = V^T rows (contiguous global) ---
    v16h Vb[4];
#pragma unroll
    for (int es = 0; es < 4; ++es) {
      const _Float16* r = vp + (size_t)(es * 16 + lm) * Sq + t0 + kbB;
      union { v16h v; v4i q[2]; } u;
      u.q[0] = *(const v4i*)r;
      u.q[1] = *(const v4i*)(r + 16);
      Vb[es] = u.v;
    }
#pragma unroll
    for (int ms = 0; ms < 2; ++ms) {
      const _Float16* pr = Pw + ms * 640 + lm * 40;
      union { v16h v; v4i q[2]; } u;
      u.q[0] = *(const v4i*)(pr + kbA);
      u.q[1] = *(const v4i*)(pr + 16 + kbA);
      const v16h Pa = u.v;
#pragma unroll
      for (int es = 0; es < 4; ++es) O[ms][es] = WMMA_F16(Pa, Vb[es], O[ms][es]);
    }
  }

  // normalize and store concat-head output [B,S,D] f16
#pragma unroll
  for (int ms = 0; ms < 2; ++ms) {
    const float linv = 1.0f / Lrow[ms];
#pragma unroll
    for (int i = 0; i < 8; ++i) {
      const float li = __shfl(linv, i + (hib ? 8 : 0), 32);
      const int s = sRow + ms * 16 + i + (hib ? 8 : 0);
      const size_t rowoff = ((size_t)b * Sq + s) * Dm + h * HSz;
#pragma unroll
      for (int es = 0; es < 4; ++es)
        og[rowoff + es * 16 + lm] = (_Float16)(O[ms][es][i] * li);
    }
  }
}

// ---------------------------------------------------------------------------
// Host-side orchestration
// ---------------------------------------------------------------------------
extern "C" void kernel_launch(void* const* d_in, const int* in_sizes, int n_in,
                              void* d_out, int out_size, void* d_ws, size_t ws_size,
                              hipStream_t stream) {
  const float* x     = (const float*)d_in[0];
  const float* Wq    = (const float*)d_in[1];
  const float* Wk    = (const float*)d_in[2];
  const float* Wv    = (const float*)d_in[3];
  const float* Wproj = (const float*)d_in[4];
  const float* bproj = (const float*)d_in[5];
  const float* ln1g  = (const float*)d_in[6];
  const float* ln1b  = (const float*)d_in[7];
  const float* ln2g  = (const float*)d_in[8];
  const float* ln2b  = (const float*)d_in[9];
  const float* W1    = (const float*)d_in[10];
  const float* b1    = (const float*)d_in[11];
  const float* W2    = (const float*)d_in[12];
  const float* b2    = (const float*)d_in[13];
  float* out = (float*)d_out;

  char* ws = (char*)d_ws;
  const size_t MB = 1024 * 1024;
  _Float16* h16    = (_Float16*)(ws + 0);          // 8 MB  [4096,1024]
  _Float16* WqT    = (_Float16*)(ws + 8  * MB);    // 2 MB  [1024,1024]
  _Float16* WkT    = (_Float16*)(ws + 10 * MB);
  _Float16* WvT    = (_Float16*)(ws + 12 * MB);
  _Float16* WprojT = (_Float16*)(ws + 14 * MB);
  _Float16* W1T    = (_Float16*)(ws + 16 * MB);    // 8 MB  [4096,1024]
  _Float16* W2T    = (_Float16*)(ws + 24 * MB);    // 8 MB  [1024,4096]
  _Float16* q16    = (_Float16*)(ws + 32 * MB);    // 8 MB  [B,H,S,HS]
  _Float16* k16    = (_Float16*)(ws + 40 * MB);    // 8 MB
  _Float16* vT16   = (_Float16*)(ws + 48 * MB);    // 8 MB  [B,H,HS,S]
  _Float16* o16    = (_Float16*)(ws + 56 * MB);    // 8 MB  [4096,1024]
  _Float16* h2_16  = h16;                          // alias: h1 dead after QKV
  _Float16* act16  = q16;                          // alias: q/k/v/o dead after proj (32 MB)

  const int M = Bsz * Sq;  // 4096

  // Weight conversions (fp32 -> f16 transposed [N][K])
  cvt_qkvT<<<4096, 256, 0, stream>>>(Wq, WqT);
  cvt_qkvT<<<4096, 256, 0, stream>>>(Wk, WkT);
  cvt_qkvT<<<4096, 256, 0, stream>>>(Wv, WvT);
  cvtT<<<(Dm * Dm) / 256, 256, 0, stream>>>(Wproj, WprojT, Dm, Dm);
  cvtT<<<(Dm * 4 * Dm) / 256, 256, 0, stream>>>(W1, W1T, Dm, 4 * Dm);
  cvtT<<<(4 * Dm * Dm) / 256, 256, 0, stream>>>(W2, W2T, 4 * Dm, Dm);

  // LN1
  ln_kernel<<<M, 256, 0, stream>>>(x, ln1g, ln1b, h16);

  // QKV projections (scale 1/sqrt(HS)=0.125 folded into q)
  dim3 gD(Dm / 128, M / 128);  // (8, 32)
  gemm_wmma<EPI_Q><<<gD, 256, 0, stream>>>(h16, WqT, q16, nullptr, nullptr, nullptr, M, Dm, Dm, 0.125f);
  gemm_wmma<EPI_K><<<gD, 256, 0, stream>>>(h16, WkT, k16, nullptr, nullptr, nullptr, M, Dm, Dm, 1.0f);
  gemm_wmma<EPI_V><<<gD, 256, 0, stream>>>(h16, WvT, vT16, nullptr, nullptr, nullptr, M, Dm, Dm, 1.0f);

  // Attention (full, no mask), concat-head output o16
  attn_wmma<<<dim3(Sq / 256, Bsz * Hh), 256, 0, stream>>>(q16, k16, vT16, o16);

  // Output projection + residual -> d_out (fp32)
  gemm_wmma<EPI_PROJ><<<gD, 256, 0, stream>>>(o16, WprojT, nullptr, out, bproj, x, M, Dm, Dm, 1.0f);

  // LN2
  ln_kernel<<<M, 256, 0, stream>>>(out, ln2g, ln2b, h2_16);

  // MLP
  dim3 gF1(4 * Dm / 128, M / 128);  // (32, 32)
  gemm_wmma<EPI_FF1><<<gF1, 256, 0, stream>>>(h2_16, W1T, act16, nullptr, b1, nullptr, M, 4 * Dm, Dm, 1.0f);
  gemm_wmma<EPI_FF2><<<gD, 256, 0, stream>>>(act16, W2T, nullptr, out, b2, out, M, Dm, 4 * Dm, 1.0f);
}
